// MLPPredictor_41154376630436
// MI455X (gfx1250) — matrix operand order repaired and checked by reading, not compiled, not run on hardware
//
#include <hip/hip_runtime.h>
#include <hip/hip_bf16.h>

typedef __attribute__((ext_vector_type(16))) __bf16 v16bf;
typedef __attribute__((ext_vector_type(8)))  __bf16 v8bf;
typedef __attribute__((ext_vector_type(8)))  float  v8f;

#define HIDDEN 128
#define KDIM   256            // 2*HIDDEN (concat of src/dst features)
#define MTILES 4              // 16-row M tiles per block iteration
#define TILE_M (16 * MTILES)  // 64 edges per block iteration
#define KSTEPS 8              // 256 / 32
#define AROW   (KDIM + 8)     // bf16 row stride: 528B = 132 dwords -> 4-bank step
#define YROW   (HIDDEN + 4)   // fp32 row stride: 528B
#define SMEM_BYTES (TILE_M * AROW * 2)   // == TILE_M * YROW * 4 == 33792

// Hard scheduling fence: nothing may be reordered across it.
#define SB() __builtin_amdgcn_sched_barrier(0)

// Load one 16x32 bf16 A fragment (two 16B LDS reads) for tile t, k-step ks.
#define LOAD_A(dstv, t, ks)                                              \
    {                                                                    \
        const __bf16* ap_ = arow + (t) * (16 * AROW) + (ks) * 32;        \
        v8bf lo_ = *(const v8bf*)(ap_);                                  \
        v8bf hi_ = *(const v8bf*)(ap_ + 16);                             \
        v16bf av_;                                                       \
        _Pragma("unroll")                                                \
        for (int j_ = 0; j_ < 8; ++j_) { av_[j_] = lo_[j_]; av_[8 + j_] = hi_[j_]; } \
        dstv = av_;                                                      \
    }

// Load the 4 M-tile fragments of one k-step into buffer set P (P0..P3).
#define LOAD4(P, ks)                                                     \
    LOAD_A(P##0, 0, ks); LOAD_A(P##1, 1, ks);                            \
    LOAD_A(P##2, 2, ks); LOAD_A(P##3, 3, ks);

// Issue the 4 independent WMMAs of one k-step from buffer set P.
#define WMMA4(P, ks)                                                     \
    c0 = __builtin_amdgcn_wmma_f32_16x16x32_bf16(false, P##0, false, Bfrag[ks], \
                                                 (short)0, c0, false, false);   \
    c1 = __builtin_amdgcn_wmma_f32_16x16x32_bf16(false, P##1, false, Bfrag[ks], \
                                                 (short)0, c1, false, false);   \
    c2 = __builtin_amdgcn_wmma_f32_16x16x32_bf16(false, P##2, false, Bfrag[ks], \
                                                 (short)0, c2, false, false);   \
    c3 = __builtin_amdgcn_wmma_f32_16x16x32_bf16(false, P##3, false, Bfrag[ks], \
                                                 (short)0, c3, false, false);

__global__ __launch_bounds__(256)
void edge_mlp_wmma_kernel(const float* __restrict__ h,
                          const int*   __restrict__ src,
                          const int*   __restrict__ dst,
                          const float* __restrict__ W1,
                          const float* __restrict__ b1,
                          const float* __restrict__ gamma,
                          const float* __restrict__ beta,
                          const float* __restrict__ mean,
                          const float* __restrict__ var,
                          const float* __restrict__ W2,
                          const float* __restrict__ b2,
                          float*       __restrict__ out,
                          int n_edges)
{
    // A tile (bf16, gathered) and Y tile (fp32, post-BN) alias the same LDS:
    // both are exactly 33792 bytes, lifetimes separated by barriers.
    __shared__ __align__(16) unsigned char Smem[SMEM_BYTES];
    __shared__ __align__(16) float W2lds[2 * HIDDEN];      // 1024 B

    __bf16* Alds = (__bf16*)Smem;
    float*  Ylds = (float*)Smem;

    const int tid   = threadIdx.x;
    const int wave  = tid >> 5;          // 0..7 -> owns N columns [wave*16, wave*16+16)
    const int lane  = tid & 31;
    const int lmod  = lane & 15;
    const int half  = lane >> 4;         // hi/lo half of wave (per WMMA lane layout)
    const int nlane = wave * 16 + lmod;  // this lane's output-feature column (0..127)

    // Stage W2 (2x128 fp32) once.
    W2lds[tid] = W2[tid];

    // ---- Preload B fragments (W1^T slice for this wave) into registers, bf16 ----
    // B lane layout (16-bit 32x16): lane element j corresponds to K = ks*32 + {half*8+0..7, 16+half*8+0..7},
    // i.e. two 8-element K runs of W1 row `nlane`.
    v16bf Bfrag[KSTEPS];
#pragma unroll
    for (int ks = 0; ks < KSTEPS; ++ks) {
        const float* wp = W1 + (size_t)nlane * KDIM + ks * 32 + half * 8;
        v16bf b;
#pragma unroll
        for (int j = 0; j < 16; ++j) b[j] = (__bf16)wp[(j < 8) ? j : j + 8];
        Bfrag[ks] = b;
    }

    // ---- Fold BatchNorm (eval) into per-column scale/shift, once per lane ----
    const float sc    = gamma[nlane] * rsqrtf(var[nlane] + 1e-5f);
    const float sh    = beta[nlane] - mean[nlane] * sc;
    const float bias1 = b1[nlane];

    const int numTiles = n_edges / TILE_M;
    for (int tile = blockIdx.x; tile < numTiles; tile += gridDim.x) {
        const int base = tile * TILE_M;

        // Prefetch next iteration's edge indices into cache.
        const int nt = tile + gridDim.x;
        if (nt < numTiles) {
            __builtin_prefetch(src + nt * TILE_M, 0, 3);
            __builtin_prefetch(dst + nt * TILE_M, 0, 3);
        }

        // ---- Gather + convert: A[e][0:128]=h[src[e]], A[e][128:256]=h[dst[e]] ----
        // 64 edges x 16 chunks = 1024 chunk tasks over 256 threads (4 each).
#pragma unroll
        for (int i = 0; i < MTILES; ++i) {
            const int cidx = tid + i * 256;
            const int e  = cidx >> 4;            // edge within tile (0..63)
            const int p  = cidx & 15;            // 16-element chunk (0..15)
            const int node = (p < 8) ? src[base + e] : dst[base + e];
            const float* hp = h + (size_t)node * HIDDEN + (p & 7) * 16;
            __bf16* ap = Alds + e * AROW + p * 16;
#pragma unroll
            for (int j = 0; j < 16; j += 4) {
                float4 f = *(const float4*)(hp + j);
                ap[j + 0] = (__bf16)f.x;
                ap[j + 1] = (__bf16)f.y;
                ap[j + 2] = (__bf16)f.z;
                ap[j + 3] = (__bf16)f.w;
            }
        }
        __syncthreads();

        // ---- WMMA main loop: pinned ping-pong pipeline (sched_barrier fenced) ----
        // A lane layout (16-bit 16x32): lane m = lane&15 holds row m;
        // per k-step its 16 elements are K = ks*32 + {half*8+0..7, 16+half*8+0..7}.
        v8f c0 = {0.f,0.f,0.f,0.f,0.f,0.f,0.f,0.f};
        v8f c1 = c0, c2 = c0, c3 = c0;
        const __bf16* arow = Alds + lmod * AROW + half * 8;

        v16bf A0_0, A0_1, A0_2, A0_3;    // even k-step buffer set
        v16bf A1_0, A1_1, A1_2, A1_3;    // odd  k-step buffer set

        LOAD4(A0_, 0);                   // 8 loads in flight
        LOAD4(A1_, 1);                   // 16 loads in flight
        SB();
        WMMA4(A0_, 0);                   // waits only for first 8 loads
        SB();
        LOAD4(A0_, 2);
        SB();
        WMMA4(A1_, 1);
        SB();
        LOAD4(A1_, 3);
        SB();
        WMMA4(A0_, 2);
        SB();
        LOAD4(A0_, 4);
        SB();
        WMMA4(A1_, 3);
        SB();
        LOAD4(A1_, 5);
        SB();
        WMMA4(A0_, 4);
        SB();
        LOAD4(A0_, 6);
        SB();
        WMMA4(A1_, 5);
        SB();
        LOAD4(A1_, 7);
        SB();
        WMMA4(A0_, 6);
        SB();
        WMMA4(A1_, 7);
        SB();

        __syncthreads();   // all A reads done before Y (aliased) is written

        // ---- Epilogue: bias + ReLU + BN -> Ylds (aliases Alds) ----
        // C/D layout: lane element r -> (m = r + 8*half, n = nlane)
        {
            v8f cs[MTILES] = {c0, c1, c2, c3};
#pragma unroll
            for (int t = 0; t < MTILES; ++t) {
#pragma unroll
                for (int r = 0; r < 8; ++r) {
                    float v = cs[t][r] + bias1;
                    v = v > 0.f ? v : 0.f;
                    v = v * sc + sh;
                    Ylds[(t * 16 + r + 8 * half) * YROW + nlane] = v;
                }
            }
        }
        __syncthreads();

        // ---- Projection: out[e][cls] = Y[e][:] . W2[cls][:] + b2[cls] ----
        if (tid < 2 * TILE_M) {
            const int e   = tid >> 1;            // 0..63
            const int cls = tid & 1;
            float acc = b2[cls];
            const float4* yr4 = (const float4*)(Ylds + e * YROW);     // 16B aligned
            const float4* w24 = (const float4*)(W2lds + cls * HIDDEN);
#pragma unroll
            for (int k = 0; k < HIDDEN / 4; ++k) {
                float4 y = yr4[k];               // ds_load_b128
                float4 w = w24[k];               // ds_load_b128
                acc += y.x * w.x + y.y * w.y + y.z * w.z + y.w * w.w;
            }
            out[(size_t)(base + e) * 2 + cls] = acc;   // coalesced 512B burst
        }
        __syncthreads();   // Ylds consumed before next gather overwrites Smem
    }
}

// Scalar tail for E % 64 edges (E = 1M is divisible; kept for determinism/safety).
__global__ void edge_mlp_tail_kernel(const float* __restrict__ h,
                                     const int*   __restrict__ src,
                                     const int*   __restrict__ dst,
                                     const float* __restrict__ W1,
                                     const float* __restrict__ b1,
                                     const float* __restrict__ gamma,
                                     const float* __restrict__ beta,
                                     const float* __restrict__ mean,
                                     const float* __restrict__ var,
                                     const float* __restrict__ W2,
                                     const float* __restrict__ b2,
                                     float*       __restrict__ out,
                                     int start, int n_edges)
{
    const int e = start + blockIdx.x * blockDim.x + threadIdx.x;
    if (e >= n_edges) return;
    const float* hs = h + (size_t)src[e] * HIDDEN;
    const float* hd = h + (size_t)dst[e] * HIDDEN;
    float o0 = b2[0], o1 = b2[1];
    for (int n = 0; n < HIDDEN; ++n) {
        float acc = b1[n];
        const float* w = W1 + (size_t)n * KDIM;
        for (int k = 0; k < HIDDEN; ++k) acc += hs[k] * w[k] + hd[k] * w[HIDDEN + k];
        acc = acc > 0.f ? acc : 0.f;
        const float s = gamma[n] * rsqrtf(var[n] + 1e-5f);
        acc = (acc - mean[n]) * s + beta[n];
        o0 += acc * W2[n];
        o1 += acc * W2[HIDDEN + n];
    }
    out[(size_t)e * 2 + 0] = o0;
    out[(size_t)e * 2 + 1] = o1;
}

extern "C" void kernel_launch(void* const* d_in, const int* in_sizes, int n_in,
                              void* d_out, int out_size, void* d_ws, size_t ws_size,
                              hipStream_t stream) {
    const float* h     = (const float*)d_in[0];
    const int*   src   = (const int*)  d_in[1];
    const int*   dst   = (const int*)  d_in[2];
    const float* W1    = (const float*)d_in[3];
    const float* b1    = (const float*)d_in[4];
    const float* gamma = (const float*)d_in[5];
    const float* beta  = (const float*)d_in[6];
    const float* mean  = (const float*)d_in[7];
    const float* var   = (const float*)d_in[8];
    const float* W2    = (const float*)d_in[9];
    const float* b2    = (const float*)d_in[10];
    float* out = (float*)d_out;

    const int E = in_sizes[1];
    const int numTiles = E / TILE_M;

    if (numTiles > 0) {
        int blocks = numTiles < 2048 ? numTiles : 2048;
        edge_mlp_wmma_kernel<<<blocks, 256, 0, stream>>>(
            h, src, dst, W1, b1, gamma, beta, mean, var, W2, b2, out, E);
    }
    const int tail = E - numTiles * TILE_M;
    if (tail > 0) {
        const int tblocks = (tail + 63) / 64;
        edge_mlp_tail_kernel<<<tblocks, 64, 0, stream>>>(
            h, src, dst, W1, b1, gamma, beta, mean, var, W2, b2, out,
            numTiles * TILE_M, E);
    }
}
